// AttnBlock_43508018708507
// MI455X (gfx1250) — matrix operand order repaired and checked by reading, not compiled, not run on hardware
//
#include <hip/hip_runtime.h>
#include <hip/hip_bf16.h>

typedef __attribute__((ext_vector_type(16))) __bf16 v16bf;
typedef __attribute__((ext_vector_type(8)))  float  v8f;
typedef __attribute__((ext_vector_type(4)))  float  fvec4;   // clang vector: OK for NT builtins

#define DIM    512
#define HEADS  8
#define HD     64           // head dim
#define MROWS  8192         // 8*32*32 pixels
#define BN     64           // b*n pairs
#define HW     1024         // 32*32 pixels per image

union ABFrag { v16bf v; uint4 u[2]; };

__device__ __forceinline__ unsigned bfbits(float f) {
    unsigned u = __float_as_uint(f);
    return (u + 0x7FFFu + ((u >> 16) & 1u)) >> 16;   // round-to-nearest-even bf16
}

// ---------------- f32 -> bf16 conversion (2 elems/thread, packed u32 stores) ----
__global__ void __launch_bounds__(256)
cvt_bf16_kernel(const float* __restrict__ in, unsigned* __restrict__ out, int n2) {
    int i = blockIdx.x * blockDim.x + threadIdx.x;
    if (i >= n2) return;
    float2 f = ((const float2*)in)[i];
    out[i] = (bfbits(f.y) << 16) | bfbits(f.x);
}

// ---------------- fused QKV projection (WMMA) + per-row head reductions --------
// grid = 512 blocks (16 rows each), block = 256 threads = 8 waves (1 wave per head).
// Wave computes q/k/v tiles [16 x 64] via v_wmma_f32_16x16x32_bf16 over K=512,
// then reduces: logits[row,head] = sum_d q*k ; vsum[row,head] = sum_d v.
__global__ void __launch_bounds__(256)
qkv_reduce_kernel(const unsigned short* __restrict__ xb,
                  const unsigned short* __restrict__ wqb,
                  const unsigned short* __restrict__ wkb,
                  const unsigned short* __restrict__ wvb,
                  float* __restrict__ logits, float* __restrict__ vsum) {
    const int lane   = threadIdx.x & 31;
    const int head   = threadIdx.x >> 5;          // 0..7
    const int laneLo = lane & 15;
    const int hiHalf = lane >> 4;                 // 0 or 1
    const int m0     = blockIdx.x * 16;

    // A: 16x32, contiguous k order used here (same as B). lanes<16: elems 0..15 = K0..15
    //                          lanes>=16: K16..31
    const int aSel = hiHalf * 16;
    // B: 32x16, lane = column; lanes<16 hold K0..15, lanes>=16 hold K16..31
    const int bSel = hiHalf * 16;

    const unsigned short* arow = xb + (size_t)(m0 + laneLo) * DIM;
    const int col = head * HD + laneLo;           // base column (per 16-wide N tile +16*j)

    v8f qa[4], ka[4], va[4];
#pragma unroll
    for (int j = 0; j < 4; ++j) { qa[j] = (v8f){}; ka[j] = (v8f){}; va[j] = (v8f){}; }

    for (int kt = 0; kt < DIM; kt += 32) {
        ABFrag a;
        a.u[0] = *(const uint4*)(arow + kt + aSel);
        a.u[1] = *(const uint4*)(arow + kt + aSel + 8);
#pragma unroll
        for (int j = 0; j < 4; ++j) {
            const size_t wro = (size_t)(col + 16 * j) * DIM + kt + bSel;
            ABFrag bq, bk, bv;
            bq.u[0] = *(const uint4*)(wqb + wro);
            bq.u[1] = *(const uint4*)(wqb + wro + 8);
            qa[j] = __builtin_amdgcn_wmma_f32_16x16x32_bf16(false, a.v, false, bq.v,
                                                            (short)0, qa[j], false, false);
            bk.u[0] = *(const uint4*)(wkb + wro);
            bk.u[1] = *(const uint4*)(wkb + wro + 8);
            ka[j] = __builtin_amdgcn_wmma_f32_16x16x32_bf16(false, a.v, false, bk.v,
                                                            (short)0, ka[j], false, false);
            bv.u[0] = *(const uint4*)(wvb + wro);
            bv.u[1] = *(const uint4*)(wvb + wro + 8);
            va[j] = __builtin_amdgcn_wmma_f32_16x16x32_bf16(false, a.v, false, bv.v,
                                                            (short)0, va[j], false, false);
        }
    }

    // C layout: VGPR i -> (M=i, N=lane) lanes 0..15 ; (M=i+8, N=lane-16) lanes 16..31.
    // Reduce q*k and v over all 64 columns (4 tiles x 16 lanes of one half-wave).
    const int b  = m0 >> 10;
    const int h  = (m0 >> 5) & 31;
    const int w0 = m0 & 31;
#pragma unroll
    for (int i = 0; i < 8; ++i) {
        float r = qa[0][i] * ka[0][i] + qa[1][i] * ka[1][i] +
                  qa[2][i] * ka[2][i] + qa[3][i] * ka[3][i];
        float u = va[0][i] + va[1][i] + va[2][i] + va[3][i];
#pragma unroll
        for (int m = 1; m < 16; m <<= 1) {          // stays within each 16-lane half
            r += __shfl_xor(r, m, 32);
            u += __shfl_xor(u, m, 32);
        }
        if (laneLo == 0) {
            const int row = i + hiHalf * 8;          // tile-local row -> w
            const size_t idx = (((size_t)b * HEADS + head) * 32 + h) * 32 + (w0 + row);
            logits[idx] = r;
            vsum[idx]   = u;
        }
    }
}

// ---------------- softmax over w (rows of 32), scale = d^-0.5 = 0.125 ----------
__global__ void __launch_bounds__(256)
softmax_kernel(float* __restrict__ logits, int nrows) {
    const int gid  = blockIdx.x * blockDim.x + threadIdx.x;
    const int row  = gid >> 5;
    const int lane = gid & 31;
    if (row >= nrows) return;
    float v = logits[row * 32 + lane] * 0.125f;
    float m = v;
#pragma unroll
    for (int s = 1; s < 32; s <<= 1) m = fmaxf(m, __shfl_xor(m, s, 32));
    float e = __expf(v - m);
    float sum = e;
#pragma unroll
    for (int s = 1; s < 32; s <<= 1) sum += __shfl_xor(sum, s, 32);
    logits[row * 32 + lane] = e / sum;
}

// ---------------- outer product + streaming NT stores (256 MB) -----------------
// out[((bn*1024 + hw)*1024) + xy] = W[bn,hw] * S[bn,xy]
// grid = 64 (bn) * 64 (chunks of 16 hw) ; block = 256 threads ; fvec4 per thread.
__global__ void __launch_bounds__(256)
outer_kernel(const float* __restrict__ wgt, const float* __restrict__ svec,
             float* __restrict__ out) {
    const int bn    = blockIdx.x >> 6;
    const int chunk = blockIdx.x & 63;
    const int tid   = threadIdx.x;
    const fvec4 sv = ((const fvec4*)(svec + bn * HW))[tid];   // held in registers
    const int hw0   = chunk * 16;
    const float* wrow = wgt + bn * HW + hw0;
    fvec4* obase = (fvec4*)out + ((size_t)bn * HW + hw0) * 256 + tid;
#pragma unroll 4
    for (int i = 0; i < 16; ++i) {
        const float ws = wrow[i];
        fvec4 o = sv * ws;
        __builtin_nontemporal_store(o, obase);   // NT: 256MB stream > 192MB L2
        obase += 256;
    }
}

extern "C" void kernel_launch(void* const* d_in, const int* in_sizes, int n_in,
                              void* d_out, int out_size, void* d_ws, size_t ws_size,
                              hipStream_t stream) {
    const float* x  = (const float*)d_in[0];
    const float* wq = (const float*)d_in[1];
    const float* wk = (const float*)d_in[2];
    const float* wv = (const float*)d_in[3];
    float* out = (float*)d_out;

    char* ws = (char*)d_ws;
    unsigned short* xb  = (unsigned short*)ws;                       // 8 MB
    unsigned short* wqb = (unsigned short*)(ws + (size_t)(8 << 20));
    unsigned short* wkb = wqb + DIM * DIM;
    unsigned short* wvb = wkb + DIM * DIM;
    float* logits = (float*)(ws + (size_t)(8 << 20) + (size_t)3 * DIM * DIM * 2);
    float* vsum   = logits + BN * HW;                                // 256 KB each

    // bf16 conversions
    {
        int n2 = MROWS * DIM / 2;
        cvt_bf16_kernel<<<(n2 + 255) / 256, 256, 0, stream>>>(x, (unsigned*)xb, n2);
        int w2 = DIM * DIM / 2;
        cvt_bf16_kernel<<<(w2 + 255) / 256, 256, 0, stream>>>(wq, (unsigned*)wqb, w2);
        cvt_bf16_kernel<<<(w2 + 255) / 256, 256, 0, stream>>>(wk, (unsigned*)wkb, w2);
        cvt_bf16_kernel<<<(w2 + 255) / 256, 256, 0, stream>>>(wv, (unsigned*)wvb, w2);
    }

    // fused QKV projection + reductions (WMMA)
    qkv_reduce_kernel<<<MROWS / 16, 256, 0, stream>>>(xb, wqb, wkb, wvb, logits, vsum);

    // softmax over w
    softmax_kernel<<<(BN * 32 * 32 + 255) / 256, 256, 0, stream>>>(logits, BN * 32);

    // outer product -> 256 MB streaming output
    outer_kernel<<<BN * 64, 256, 0, stream>>>(logits, vsum, out);
}
